// BaichuanAttention_43044162240761
// MI455X (gfx1250) — matrix operand order repaired and checked by reading, not compiled, not run on hardware
//
#include <hip/hip_runtime.h>

// ---------------------------------------------------------------------------
// Types for CDNA5 WMMA (wave32): v_wmma_f32_16x16x32_bf16
// ---------------------------------------------------------------------------
typedef __attribute__((ext_vector_type(16))) __bf16 v16bf;
typedef __attribute__((ext_vector_type(8)))  float  v8f;

#define NUM_HEADS 32
#define HEAD_DIM  128
#define SEQ       2048
#define HID       4096
#define HID3      12288

// ---------------------------------------------------------------------------
// Tensor Data Mover (gfx1250). Guarded: falls back to manual staging if the
// builtin is absent. Arity differs between toolchains (5-arg ROCm clang-22,
// 6-arg amdgpu-toolchain clang-23 whose TDM header we can detect).
// ---------------------------------------------------------------------------
#if defined(__gfx1250__) && __has_builtin(__builtin_amdgcn_tensor_load_to_lds)
#define USE_TDM 1
#if __has_include(<hip/amd_detail/amd_gfx1250_TDM.h>)
#define BAICHUAN_TDM_6ARG 1
#endif
typedef __attribute__((ext_vector_type(4))) unsigned int tdm_u32x4;
typedef __attribute__((ext_vector_type(8))) int          tdm_i32x8;
typedef __attribute__((ext_vector_type(4))) int          tdm_i32x4;

// 2D tile load, 2-byte elements. Optional LDS pad: 1 dword per 64 dwords
// (i.e. +2 ushorts after each 128-element row -> row stride 130).
__device__ __forceinline__ void tdm_load_2d(const void* gptr,
                                            unsigned lds_byte_addr,
                                            unsigned tensor_d0, unsigned tensor_d1,
                                            unsigned tile_d0, unsigned tile_d1,
                                            unsigned stride0, bool pad_row128) {
    unsigned long long ga = (unsigned long long)(uintptr_t)gptr;
    tdm_u32x4 g0;
    g0.x = 1u;                                        // count=1 (valid D#)
    g0.y = lds_byte_addr;                             // LDS dest (bytes)
    g0.z = (unsigned)(ga & 0xffffffffu);              // global_addr[31:0]
    g0.w = (unsigned)((ga >> 32) & 0x01ffffffu)       // global_addr[56:32]
         | (2u << 30);                                // type=2 ("image")
    unsigned w0 = (1u << 16);                         // data_size=1 -> 2 bytes
    if (pad_row128) w0 |= (1u << 20)                  // pad_enable
                       |  (5u << 22)                  // pad_interval: 64 dwords
                       |  (0u << 25);                 // pad_amount: 1 dword
    tdm_i32x8 g1;
    g1[0] = (int)w0;                                  // mask=0, ds=2B, pads
    g1[1] = (int)((tensor_d0 & 0xffffu) << 16);       // tensor_dim0 lo16
    g1[2] = (int)(((tensor_d0 >> 16) & 0xffffu) | ((tensor_d1 & 0xffffu) << 16));
    g1[3] = (int)(((tensor_d1 >> 16) & 0xffffu) | ((tile_d0 & 0xffffu) << 16));
    g1[4] = (int)(tile_d1 & 0xffffu);                 // tile_dim1, tile_dim2=0
    g1[5] = (int)stride0;                             // tensor_dim0_stride lo32
    g1[6] = 0;
    g1[7] = 0;
    tdm_i32x4 z4 = {0, 0, 0, 0};
#if defined(BAICHUAN_TDM_6ARG)
    tdm_i32x8 z8 = {0, 0, 0, 0, 0, 0, 0, 0};
    __builtin_amdgcn_tensor_load_to_lds(g0, g1, z4, z4, z8, 0);
#else
    __builtin_amdgcn_tensor_load_to_lds(g0, g1, z4, z4, 0);
#endif
}
#endif // USE_TDM

__device__ __forceinline__ unsigned short f32_to_bf16(float f) {
    unsigned int u = __float_as_uint(f);
    unsigned int lsb = (u >> 16) & 1u;
    u += 0x7fffu + lsb;                      // round-to-nearest-even
    return (unsigned short)(u >> 16);
}

// ---------------------------------------------------------------------------
// Fragment loaders (CDNA5 16-bit WMMA VGPR layouts, wave32).
// A (16x32, MxK): lane L holds row M=L%16; h=L/16 selects K-interleave:
//   VGPR v in 0..3 -> K = 8h+2v,+1 ; v in 4..7 -> K = 16+8h+2(v-4),+1
// B (32x16, KxN) stored k-major in LDS: lane L holds N=L%16; h=L/16, same K-interleave as A
// C/D (16x16): lane L holds N=L%16; VGPR r -> M = r + 8*(L/16)
// ---------------------------------------------------------------------------
__device__ __forceinline__ v16bf load_a_frag(const unsigned short* base, int ld,
                                             int mbase, int kbase, int lane) {
    int m = mbase + (lane & 15);
    int h = (lane >> 4) & 1;
    const unsigned short* row = base + (size_t)m * ld + kbase;
    union { unsigned int u[8]; v16bf v; } f;
#pragma unroll
    for (int j = 0; j < 4; ++j) {
        f.u[j]     = *(const unsigned int*)(row + 8 * h + 2 * j);
        f.u[j + 4] = *(const unsigned int*)(row + 16 + 8 * h + 2 * j);
    }
    return f.v;
}

__device__ __forceinline__ v16bf load_b_frag(const unsigned short* base, int ld,
                                             int krow, int nbase) {
    const int l = krow & 31;
    const int hh = l >> 4;
    const unsigned short* col = base + (size_t)(krow - l) * ld + nbase + (l & 15);
    union { unsigned short s[16]; v16bf v; } f;
#pragma unroll
    for (int j = 0; j < 8; ++j) {
        f.s[j]     = col[(size_t)(8 * hh + j) * ld];
        f.s[j + 8] = col[(size_t)(16 + 8 * hh + j) * ld];
    }
    return f.v;
}

// ---------------------------------------------------------------------------
// fp32 -> bf16 conversion (grid-stride)
// ---------------------------------------------------------------------------
__global__ void baichuan_cvt_bf16(const float* __restrict__ in,
                                  unsigned short* __restrict__ out, int n) {
    int i = blockIdx.x * blockDim.x + threadIdx.x;
    int stride = gridDim.x * blockDim.x;
    for (; i < n; i += stride) out[i] = f32_to_bf16(in[i]);
}

// ---------------------------------------------------------------------------
// Tiled bf16 GEMM:  C[M,N] = A[M,K] @ B[N,K]^T   (B is weight, row-major NxK)
// Block: 256 threads = 8 waves. Block tile 128x128, wave tile 32x64,
// k-step 32 (one WMMA K). A tile staged via TDM when available.
// ---------------------------------------------------------------------------
__global__ void __launch_bounds__(256)
baichuan_gemm_bf16(const unsigned short* __restrict__ A,
                   const unsigned short* __restrict__ B,
                   float* __restrict__ Cf, unsigned short* __restrict__ Cb,
                   int M, int N, int K) {
    __shared__ unsigned short Alds[128][32];        // row-major  (8 KB)
    __shared__ unsigned short Blds[32][130];        // k-major +pad (8.1 KB)

    const int tid  = threadIdx.x;
    const int lane = tid & 31;
    const int wave = tid >> 5;
    const int bm = blockIdx.y * 128;
    const int bn = blockIdx.x * 128;
    const int wm = (wave & 3) * 32;                 // wave M offset in tile
    const int wn = (wave >> 2) * 64;                // wave N offset in tile

    v8f acc[2][4];
    const v8f zero8 = {0.f, 0.f, 0.f, 0.f, 0.f, 0.f, 0.f, 0.f};
#pragma unroll
    for (int mi = 0; mi < 2; ++mi)
#pragma unroll
        for (int ni = 0; ni < 4; ++ni) acc[mi][ni] = zero8;

    const int r  = tid >> 1;                        // 0..127
    const int c  = (tid & 1) * 16;                  // 0 / 16
#if USE_TDM
    const unsigned alds_addr = (unsigned)(uintptr_t)&Alds[0][0];
#endif

    for (int k0 = 0; k0 < K; k0 += 32) {
        // --- stage A tile 128x32 ---
#if USE_TDM
        if (wave == 0) {
            tdm_load_2d(A + (size_t)bm * K + k0, alds_addr,
                        (unsigned)K, (unsigned)M, 32u, 128u, (unsigned)K, false);
            __builtin_amdgcn_s_wait_tensorcnt(0);
        }
#else
        {
            const unsigned short* ag = A + (size_t)(bm + r) * K + k0 + c;
            *(uint4*)&Alds[r][c]     = ((const uint4*)ag)[0];
            *(uint4*)&Alds[r][c + 8] = ((const uint4*)ag)[1];
        }
#endif
        // --- stage B tile, transposed to k-major ---
        const unsigned short* bg = B + (size_t)(bn + r) * K + k0 + c;
        union { uint4 q[2]; unsigned short s[16]; } ub;
        ub.q[0] = ((const uint4*)bg)[0];
        ub.q[1] = ((const uint4*)bg)[1];
#pragma unroll
        for (int i = 0; i < 16; ++i) Blds[c + i][r] = ub.s[i];

        if (k0 + 32 < K)                            // hint next tile into L2
            __builtin_prefetch(bg + 32, 0, 1);
        __syncthreads();

        v16bf bfrag[4];
#pragma unroll
        for (int ni = 0; ni < 4; ++ni)
            bfrag[ni] = load_b_frag(&Blds[0][0], 130, lane, wn + ni * 16);
#pragma unroll
        for (int mi = 0; mi < 2; ++mi) {
            v16bf afrag = load_a_frag(&Alds[0][0], 32, wm + mi * 16, 0, lane);
#pragma unroll
            for (int ni = 0; ni < 4; ++ni)
                acc[mi][ni] = __builtin_amdgcn_wmma_f32_16x16x32_bf16(
                    false, afrag, false, bfrag[ni], (short)0, acc[mi][ni],
                    false, false);
        }
        __syncthreads();
    }

    const int h  = lane >> 4;
    const int nc = lane & 15;
#pragma unroll
    for (int mi = 0; mi < 2; ++mi)
#pragma unroll
        for (int ni = 0; ni < 4; ++ni) {
            int n = bn + wn + ni * 16 + nc;
#pragma unroll
            for (int rr = 0; rr < 8; ++rr) {
                int m = bm + wm + mi * 16 + rr + 8 * h;
                float val = acc[mi][ni][rr];
                if (Cf) Cf[(size_t)m * N + n] = val;
                else    Cb[(size_t)m * N + n] = f32_to_bf16(val);
            }
        }
}

// ---------------------------------------------------------------------------
// Flash attention (causal). Block = (head, 128 q-rows), 8 waves, each wave
// owns 16 q-rows. K-step = 32 key columns.
//   S = Q K^T * scale  -> online softmax -> O += P V
// proj layout: [SEQ][HID3], Q at col h*128, K at 4096+h*128, V at 8192+h*128.
// V tile staged by TDM (with hardware LDS row padding) when available.
// ---------------------------------------------------------------------------
__global__ void __launch_bounds__(256)
baichuan_flash_attn(const unsigned short* __restrict__ proj,
                    unsigned short* __restrict__ out_bf16) {
    __shared__ unsigned short KldsT[128][32];       // [d][n]  k-major for QK^T
    __shared__ unsigned short Vlds[32][130];        // [kk][d] +pad for PV
    __shared__ unsigned short Plds[8][16][32];      // per-wave P re-layout

    const int head = blockIdx.x;
    const int qb   = blockIdx.y;
    const int tid  = threadIdx.x;
    const int lane = tid & 31;
    const int wave = tid >> 5;
    const int qrow0 = qb * 128 + wave * 16;         // this wave's 16 q rows

    // ---- preload Q as 4 A-fragments (d split into 4 chunks of 32) ----
    v16bf qf[4];
    {
        const int m = qrow0 + (lane & 15);
        const int hh = (lane >> 4) & 1;
        const unsigned short* qrow = proj + (size_t)m * HID3 + head * HEAD_DIM;
#pragma unroll
        for (int ck = 0; ck < 4; ++ck) {
            union { unsigned int u[8]; v16bf v; } f;
#pragma unroll
            for (int j = 0; j < 4; ++j) {
                f.u[j]     = *(const unsigned int*)(qrow + ck * 32 + 8 * hh + 2 * j);
                f.u[j + 4] = *(const unsigned int*)(qrow + ck * 32 + 16 + 8 * hh + 2 * j);
            }
            qf[ck] = f.v;
        }
    }

    float row_max[8], row_sum[8];
#pragma unroll
    for (int rr = 0; rr < 8; ++rr) { row_max[rr] = -1e30f; row_sum[rr] = 0.f; }
    v8f oacc[8];
    const v8f zero8 = {0.f, 0.f, 0.f, 0.f, 0.f, 0.f, 0.f, 0.f};
#pragma unroll
    for (int dt = 0; dt < 8; ++dt) oacc[dt] = zero8;

    const float scale = 0.08838834764831845f;       // 1/sqrt(128)
    const int h  = lane >> 4;
    const int nc = lane & 15;
    const int kend = qb * 128 + 128;                // causal bound for block
#if USE_TDM
    const unsigned vlds_addr = (unsigned)(uintptr_t)&Vlds[0][0];
#endif

    for (int kc = 0; kc < kend; kc += 32) {
        // ---- stage V tile 32x128 (+pad) ----
#if USE_TDM
        if (wave == 0) {
            tdm_load_2d(proj + (size_t)kc * HID3 + 2 * HID + head * HEAD_DIM,
                        vlds_addr, (unsigned)HID3, (unsigned)SEQ,
                        (unsigned)HEAD_DIM, 32u, (unsigned)HID3,
                        true /* +1 dword pad per 128-elem row */);
            __builtin_amdgcn_s_wait_tensorcnt(0);
        }
#else
        {
            const int rr = tid >> 3;                // 0..31 key row
            const int cc = (tid & 7) * 16;          // 0..112 d offset
            const unsigned short* vg =
                proj + (size_t)(kc + rr) * HID3 + 2 * HID + head * HEAD_DIM + cc;
            *(uint4*)&Vlds[rr][cc]     = ((const uint4*)vg)[0];
            *(uint4*)&Vlds[rr][cc + 8] = ((const uint4*)vg)[1];
        }
#endif
        // ---- stage K tile transposed: 32 rows x 128 d -> [d][n] ----
        {
            const int rr = tid >> 3;                // 0..31 key row
            const int cc = (tid & 7) * 16;          // 0..112 d offset
            const unsigned short* kg =
                proj + (size_t)(kc + rr) * HID3 + HID + head * HEAD_DIM + cc;
            union { uint4 q[2]; unsigned short s[16]; } uk;
            uk.q[0] = ((const uint4*)kg)[0];
            uk.q[1] = ((const uint4*)kg)[1];
#pragma unroll
            for (int i = 0; i < 16; ++i) KldsT[cc + i][rr] = uk.s[i];
        }
        __syncthreads();

        // ---- S = Q K^T : 16x32 scores = 2 accumulators, 4 k-chunks each ----
        v8f s[2] = {zero8, zero8};
#pragma unroll
        for (int ck = 0; ck < 4; ++ck) {
#pragma unroll
            for (int nt = 0; nt < 2; ++nt) {
                v16bf kf = load_b_frag(&KldsT[0][0], 32, ck * 32 + lane, nt * 16);
                s[nt] = __builtin_amdgcn_wmma_f32_16x16x32_bf16(
                    false, qf[ck], false, kf, (short)0, s[nt], false, false);
            }
        }

        // ---- scale + causal mask + online softmax (per-row, group-of-16) ----
#pragma unroll
        for (int rr = 0; rr < 8; ++rr) {
            const int m = qrow0 + rr + 8 * h;       // global q row
            float lmax = -3.0e38f;
#pragma unroll
            for (int nt = 0; nt < 2; ++nt) {
                const int col = kc + nt * 16 + nc;  // global key col
                float val = s[nt][rr] * scale;
                if (col > m) val = -3.0e38f;
                s[nt][rr] = val;
                lmax = fmaxf(lmax, val);
            }
#pragma unroll
            for (int off = 1; off < 16; off <<= 1)
                lmax = fmaxf(lmax, __shfl_xor(lmax, off, 32));
            const float nm   = fmaxf(row_max[rr], lmax);
            const float corr = __expf(row_max[rr] - nm);
            float psum = 0.f;
#pragma unroll
            for (int nt = 0; nt < 2; ++nt) {
                float p = __expf(s[nt][rr] - nm);
                s[nt][rr] = p;
                psum += p;
            }
#pragma unroll
            for (int off = 1; off < 16; off <<= 1)
                psum += __shfl_xor(psum, off, 32);
            row_sum[rr] = row_sum[rr] * corr + psum;
            row_max[rr] = nm;
#pragma unroll
            for (int dt = 0; dt < 8; ++dt) oacc[dt][rr] *= corr;
        }

        // ---- restage P (C-layout) -> A-fragment layout via per-wave LDS ----
#pragma unroll
        for (int rr = 0; rr < 8; ++rr)
#pragma unroll
            for (int nt = 0; nt < 2; ++nt)
                Plds[wave][rr + 8 * h][nt * 16 + nc] = f32_to_bf16(s[nt][rr]);
        // LDS ops from the same wave are in-order (DScnt); no barrier needed.
        v16bf pf;
        {
            union { unsigned int u[8]; v16bf v; } f;
            const unsigned short* prow = &Plds[wave][lane & 15][0];
#pragma unroll
            for (int j = 0; j < 4; ++j) {
                f.u[j]     = *(const unsigned int*)(prow + 8 * h + 2 * j);
                f.u[j + 4] = *(const unsigned int*)(prow + 16 + 8 * h + 2 * j);
            }
            pf = f.v;
        }

        // ---- O += P V : 8 d-tiles of 16 ----
#pragma unroll
        for (int dt = 0; dt < 8; ++dt) {
            v16bf vf = load_b_frag(&Vlds[0][0], 130, lane, dt * 16);
            oacc[dt] = __builtin_amdgcn_wmma_f32_16x16x32_bf16(
                false, pf, false, vf, (short)0, oacc[dt], false, false);
        }
        __syncthreads();
    }

    // ---- normalize + store bf16 [SEQ][HID] ----
#pragma unroll
    for (int dt = 0; dt < 8; ++dt) {
#pragma unroll
        for (int rr = 0; rr < 8; ++rr) {
            const int m = qrow0 + rr + 8 * h;
            const int d = dt * 16 + nc;
            const float val = oacc[dt][rr] / row_sum[rr];
            out_bf16[(size_t)m * HID + head * HEAD_DIM + d] = f32_to_bf16(val);
        }
    }
}

// ---------------------------------------------------------------------------
// Launch: cvt(fp32->bf16) x3 -> GEMM(QKV, bf16 out) -> flash attn -> GEMM(out)
// ---------------------------------------------------------------------------
extern "C" void kernel_launch(void* const* d_in, const int* in_sizes, int n_in,
                              void* d_out, int out_size, void* d_ws, size_t ws_size,
                              hipStream_t stream) {
    (void)in_sizes; (void)n_in; (void)out_size; (void)ws_size;
    const float* hs = (const float*)d_in[0];
    // d_in[1] = attention_mask (pure causal; applied analytically in-kernel)
    const float* wp = (const float*)d_in[2];
    const float* wo = (const float*)d_in[3];
    float* out = (float*)d_out;

    char* ws = (char*)d_ws;
    unsigned short* hs_b   = (unsigned short*)(ws);                       // 16 MB
    unsigned short* wp_b   = (unsigned short*)(ws + (size_t)16777216);    // 96 MB
    unsigned short* wo_b   = (unsigned short*)(ws + (size_t)117440512);   // 32 MB
    unsigned short* proj_b = (unsigned short*)(ws + (size_t)150994944);   // 48 MB
    unsigned short* ao_b   = (unsigned short*)(ws + (size_t)201326592);   // 16 MB

    baichuan_cvt_bf16<<<4096, 256, 0, stream>>>(hs, hs_b, SEQ * HID);
    baichuan_cvt_bf16<<<8192, 256, 0, stream>>>(wp, wp_b, HID3 * HID);
    baichuan_cvt_bf16<<<4096, 256, 0, stream>>>(wo, wo_b, HID * HID);

    // proj[2048,12288] = hidden @ w_pack^T  (bf16 out)
    baichuan_gemm_bf16<<<dim3(HID3 / 128, SEQ / 128), 256, 0, stream>>>(
        hs_b, wp_b, nullptr, proj_b, SEQ, HID3, HID);

    // flash attention per (head, 128-row q block)
    baichuan_flash_attn<<<dim3(NUM_HEADS, SEQ / 128), 256, 0, stream>>>(
        proj_b, ao_b);

    // out[2048,4096] = attn_out @ w_o^T  (fp32 out)
    baichuan_gemm_bf16<<<dim3(HID / 128, SEQ / 128), 256, 0, stream>>>(
        ao_b, wo_b, out, nullptr, SEQ, HID, HID);
}
